// DeformConv_79319456022904
// MI455X (gfx1250) — matrix operand order repaired and checked by reading, not compiled, not run on hardware
//
#include <hip/hip_runtime.h>
#include <math.h>

typedef __attribute__((ext_vector_type(16))) __bf16 v16bf;
typedef __attribute__((ext_vector_type(8)))  __bf16 v8bf;
typedef __attribute__((ext_vector_type(8)))  float  v8f;
typedef __attribute__((ext_vector_type(4)))  unsigned int u32x4;
typedef __attribute__((ext_vector_type(8)))  int i32x8;
typedef __attribute__((ext_vector_type(4)))  int i32x4;

#define BATCH 4
#define CIN   256
#define COUT  256
#define HH    64
#define WW    64
#define KTAPS 9
#define OFFCH 18
#define HWSZ  (HH*WW)

// K2 dynamic LDS layout (bytes)
#define WROW  264              // padded weight row stride in halves (256 + 8)
#define BROW  40               // padded B row stride in halves (32 + 8)
#define SM_WHI 0               // 128*264*2 = 67584
#define SM_WLO 67584           // + 67584 = 135168
#define SM_BHI 135168          // + 64*40*2 = 5120
#define SM_BLO 140288          // + 5120 = 145408
#define SM_TAB 145408          // + 2048 = 147456
#define SMEM_BYTES 147456

// ---------------------------------------------------------------------------
// K0: transpose + bf16 hi/lo split of w_conv (O,C,9) -> wTh/wTl [9][O][C]
// ---------------------------------------------------------------------------
__global__ __launch_bounds__(256) void k0_wsplit(
    const float* __restrict__ w_conv, __bf16* __restrict__ wh,
    __bf16* __restrict__ wl)
{
    int idx = blockIdx.x * 256 + threadIdx.x;      // < 9*256*256
    int k   = idx >> 16;
    int rem = idx & 65535;
    int o   = rem >> 8;
    int c   = rem & 255;
    float v = w_conv[((size_t)o * CIN + c) * KTAPS + k];
    __bf16 hi = (__bf16)v;
    wh[idx] = hi;
    wl[idx] = (__bf16)(v - (float)hi);
}

// ---------------------------------------------------------------------------
// K1: offset conv3x3 (x -> off[B,18,H,W]). One block per (b,h) row, 64 threads.
// ---------------------------------------------------------------------------
__global__ __launch_bounds__(64) void k1_offset_conv(
    const float* __restrict__ x, const float* __restrict__ w_off,
    const float* __restrict__ b_off, float* __restrict__ off)
{
    const int bh = blockIdx.x;
    const int b  = bh >> 6;
    const int h  = bh & 63;
    const int w  = threadIdx.x;

    __shared__ float xs[16][3][66];
    __shared__ float lw[OFFCH * 16 * 9];

    float acc[OFFCH];
#pragma unroll
    for (int o = 0; o < OFFCH; ++o) acc[o] = b_off[o];

    for (int c0 = 0; c0 < CIN; c0 += 16) {
        __syncthreads();
        for (int idx = threadIdx.x; idx < OFFCH * 16 * 9; idx += 64) {
            int o   = idx / 144;
            int rem = idx - o * 144;
            int c   = rem / 9;
            int q   = rem - c * 9;
            lw[idx] = w_off[((size_t)o * CIN + (c0 + c)) * 9 + q];
        }
        for (int idx = threadIdx.x; idx < 16 * 3 * 64; idx += 64) {
            int c   = idx / 192;
            int rem = idx - c * 192;
            int r   = rem >> 6;
            int wc  = rem & 63;
            int y   = h - 1 + r;
            float v = 0.0f;
            if (y >= 0 && y < HH)
                v = x[(((size_t)b * CIN + (c0 + c)) * HH + y) * WW + wc];
            xs[c][r][wc + 1] = v;
        }
        if (threadIdx.x < 48) {
            int c = threadIdx.x / 3;
            int r = threadIdx.x % 3;
            xs[c][r][0]  = 0.0f;
            xs[c][r][65] = 0.0f;
        }
        __syncthreads();

#pragma unroll
        for (int c = 0; c < 16; ++c) {
#pragma unroll
            for (int r = 0; r < 3; ++r) {
                float x0 = xs[c][r][w + 0];
                float x1 = xs[c][r][w + 1];
                float x2 = xs[c][r][w + 2];
                const float* wp = &lw[c * 9 + r * 3];
#pragma unroll
                for (int o = 0; o < OFFCH; ++o) {
                    const float* wq = wp + o * 144;
                    acc[o] += wq[0] * x0 + wq[1] * x1 + wq[2] * x2;
                }
            }
        }
    }
#pragma unroll
    for (int o = 0; o < OFFCH; ++o)
        off[(((size_t)b * OFFCH + o) * HH + h) * WW + w] = acc[o];
}

// ---------------------------------------------------------------------------
// TDM: load a 128-row x 256-col bf16 tile (row stride 256 elems in memory)
// into LDS with 4-dword padding every 128 dwords -> LDS row stride 264 halves.
// D# built per CDNA5 ISA 8.3/8.4; forced uniform via readfirstlane.
// ---------------------------------------------------------------------------
__device__ __forceinline__ int sgpr_u(unsigned v) {
    return __builtin_amdgcn_readfirstlane((int)v);
}

__device__ __forceinline__ void tdm_load_tile_bf16(
    const void* gsrc, unsigned lds_byte_addr)
{
    unsigned long long ga = (unsigned long long)(uintptr_t)gsrc;
    u32x4 g0;
    g0[0] = (unsigned)sgpr_u(1u);                                  // count=1
    g0[1] = (unsigned)sgpr_u(lds_byte_addr);                       // lds_addr
    g0[2] = (unsigned)sgpr_u((unsigned)ga);                        // gaddr lo
    g0[3] = (unsigned)sgpr_u(((unsigned)((ga >> 32) & 0x01FFFFFFull))
                             | (2u << 30));                        // gaddr hi | type=2
    i32x8 g1;
    // data_size=2B, pad_enable, pad_interval=128 dwords (code 6),
    // pad_amount=4 dwords (code 3)
    g1[0] = sgpr_u((1u << 16) | (1u << 20) | (6u << 22) | (3u << 25));
    g1[1] = sgpr_u(256u << 16);            // tensor_dim0 = 256 (bits 63:48)
    g1[2] = sgpr_u(0u);                    // dim0 hi16 = 0, dim1 lo16 = 0
    g1[3] = sgpr_u(0x10u | (256u << 16));  // dim1 = 0x100000, tile_dim0 = 256
    g1[4] = sgpr_u(128u);                  // tile_dim1 = 128, tile_dim2 = 0
    g1[5] = sgpr_u(256u);                  // tensor_dim0_stride = 256 elems
    g1[6] = sgpr_u(0u);
    g1[7] = sgpr_u(0u);
    i32x4 z4 = {};
    i32x8 z8 = {};
    __builtin_amdgcn_tensor_load_to_lds(g0, g1, z4, z4, z8, 0);
}

// ---------------------------------------------------------------------------
// K2: fused bilinear gather + bf16-split WMMA GEMM, weights via TDM.
// Grid: (COUT/128, B*H). Block: 256 threads = 8 wave32.
// Tile: M=128 out-channels x N=64 positions; reduction 9 taps x 8 K-chunks(32).
// ---------------------------------------------------------------------------
__global__ __launch_bounds__(256) void k2_deform_gemm(
    const float* __restrict__ x, const __bf16* __restrict__ wTh,
    const __bf16* __restrict__ wTl, const float* __restrict__ off,
    float* __restrict__ pre)
{
    extern __shared__ char smem[];
    __bf16* Whi = (__bf16*)(smem + SM_WHI);
    __bf16* Wlo = (__bf16*)(smem + SM_WLO);
    __bf16* Bhi = (__bf16*)(smem + SM_BHI);
    __bf16* Blo = (__bf16*)(smem + SM_BLO);
    int*   sy0 = (int*)(smem + SM_TAB);
    int*   sx0 = sy0 + 64;
    int*   sy1 = sx0 + 64;
    int*   sx1 = sy1 + 64;
    float* sw00 = (float*)(sx1 + 64);
    float* sw01 = sw00 + 64;
    float* sw10 = sw01 + 64;
    float* sw11 = sw10 + 64;

    const int tid   = threadIdx.x;
    const int lane  = tid & 31;
    const int wave  = tid >> 5;
    const int m0    = wave * 16;
    const int Mr    = lane & 15;
    const int kb    = (lane >> 4) * 8;    // A fragment K sub-base
    const int khalf = (lane >> 4) * 8;    // B fragment K sub-base

    const int o_base = blockIdx.x * 128;
    const int bh     = blockIdx.y;
    const int b      = bh >> 6;
    const int h      = bh & 63;

    const unsigned lds_base = (unsigned)(uintptr_t)smem;

    v8f acc[4];
#pragma unroll
    for (int ns = 0; ns < 4; ++ns) { v8f zz = {}; acc[ns] = zz; }

    for (int kt = 0; kt < KTAPS; ++kt) {
        // ---- wave0 kicks off the tap's weight-tile DMAs (hi, lo planes) ----
        if (wave == 0) {
            const __bf16* gh = wTh + ((size_t)kt * COUT + o_base) * CIN;
            const __bf16* gl = wTl + ((size_t)kt * COUT + o_base) * CIN;
            tdm_load_tile_bf16(gh, lds_base + SM_WHI);
            tdm_load_tile_bf16(gl, lds_base + SM_WLO);
        }
        // ---- per-tap bilinear corner table (threads 0..63) ----
        if (tid < 64) {
            int w = tid;
            float oy = off[(((size_t)b * OFFCH + 2 * kt + 0) * HH + h) * WW + w];
            float ox = off[(((size_t)b * OFFCH + 2 * kt + 1) * HH + h) * WW + w];
            float py = (float)h + (float)(kt / 3 - 1) + oy;
            float px = (float)w + (float)(kt % 3 - 1) + ox;
            float y0f = floorf(py), x0f = floorf(px);
            float wy1 = py - y0f, wx1 = px - x0f;
            float wy0 = 1.0f - wy1, wx0 = 1.0f - wx1;
            int y0 = (int)y0f, xi0 = (int)x0f;
            int y1 = y0 + 1,  xi1 = xi0 + 1;
            float vy0 = (y0 >= 0 && y0 < HH) ? 1.0f : 0.0f;
            float vy1 = (y1 >= 0 && y1 < HH) ? 1.0f : 0.0f;
            float vx0 = (xi0 >= 0 && xi0 < WW) ? 1.0f : 0.0f;
            float vx1 = (xi1 >= 0 && xi1 < WW) ? 1.0f : 0.0f;
            sy0[w] = min(max(y0, 0), HH - 1);
            sy1[w] = min(max(y1, 0), HH - 1);
            sx0[w] = min(max(xi0, 0), WW - 1);
            sx1[w] = min(max(xi1, 0), WW - 1);
            sw00[w] = wy0 * wx0 * vy0 * vx0;
            sw01[w] = wy0 * wx1 * vy0 * vx1;
            sw10[w] = wy1 * wx0 * vy1 * vx0;
            sw11[w] = wy1 * wx1 * vy1 * vx1;
        }
        if (wave == 0) __builtin_amdgcn_s_wait_tensorcnt(0);
        __syncthreads();

        for (int cc = 0; cc < CIN; cc += 32) {
            // ---- stage B: gather + bf16 hi/lo split, 64 pos x 32 c ----
#pragma unroll
            for (int i = 0; i < 8; ++i) {
                int idx = i * 256 + tid;          // 0..2047
                int c   = idx >> 6;               // 0..31
                int p   = idx & 63;
                const float* xc = x + ((size_t)b * CIN + (cc + c)) * HWSZ;
                int ya = sy0[p], xa = sx0[p], yb = sy1[p], xb = sx1[p];
                float s = sw00[p] * xc[ya * WW + xa]
                        + sw01[p] * xc[ya * WW + xb]
                        + sw10[p] * xc[yb * WW + xa]
                        + sw11[p] * xc[yb * WW + xb];
                __bf16 hi = (__bf16)s;
                Bhi[p * BROW + c] = hi;
                Blo[p * BROW + c] = (__bf16)(s - (float)hi);
            }
            __syncthreads();

            // ---- A fragments (ISA 16-bit 16x32 layout: K {kb..kb+7, kb+16..kb+23}) ----
            const __bf16* arh = Whi + (m0 + Mr) * WROW + cc + kb;
            const __bf16* arl = Wlo + (m0 + Mr) * WROW + cc + kb;
            v8bf ah0 = *(const v8bf*)(arh);
            v8bf ah1 = *(const v8bf*)(arh + 16);
            v8bf al0 = *(const v8bf*)(arl);
            v8bf al1 = *(const v8bf*)(arl + 16);
            v16bf Ahi = __builtin_shufflevector(ah0, ah1,
                0,1,2,3,4,5,6,7,8,9,10,11,12,13,14,15);
            v16bf Alo = __builtin_shufflevector(al0, al1,
                0,1,2,3,4,5,6,7,8,9,10,11,12,13,14,15);

#pragma unroll
            for (int ns = 0; ns < 4; ++ns) {
                int p = ns * 16 + Mr;
                const __bf16* brh = Bhi + p * BROW + khalf;
                const __bf16* brl = Blo + p * BROW + khalf;
                v8bf bh0 = *(const v8bf*)(brh);
                v8bf bh1 = *(const v8bf*)(brh + 16);
                v8bf bl0 = *(const v8bf*)(brl);
                v8bf bl1 = *(const v8bf*)(brl + 16);
                v16bf Bh = __builtin_shufflevector(bh0, bh1,
                    0,1,2,3,4,5,6,7,8,9,10,11,12,13,14,15);
                v16bf Bl = __builtin_shufflevector(bl0, bl1,
                    0,1,2,3,4,5,6,7,8,9,10,11,12,13,14,15);
                acc[ns] = __builtin_amdgcn_wmma_f32_16x16x32_bf16(
                    false, Ahi, false, Bh, (short)0, acc[ns], false, false);
                acc[ns] = __builtin_amdgcn_wmma_f32_16x16x32_bf16(
                    false, Ahi, false, Bl, (short)0, acc[ns], false, false);
                acc[ns] = __builtin_amdgcn_wmma_f32_16x16x32_bf16(
                    false, Alo, false, Bh, (short)0, acc[ns], false, false);
            }
            __syncthreads();
        }
    }

    // ---- store pre-BN tile: D element (r, lane) = [M = r + 8*(lane>>4)][N = lane&15]
    const int mhi = 8 * (lane >> 4);
#pragma unroll
    for (int ns = 0; ns < 4; ++ns) {
        int wcol = ns * 16 + Mr;
#pragma unroll
        for (int r = 0; r < 8; ++r) {
            int o = o_base + m0 + mhi + r;
            pre[(((size_t)b * COUT + o) * HH + h) * WW + wcol] = acc[ns][r];
        }
    }
}

// ---------------------------------------------------------------------------
// K3: per-channel BN statistics -> (inv, shift)
// ---------------------------------------------------------------------------
__global__ __launch_bounds__(256) void k3_bn_stats(
    const float* __restrict__ pre, const float* __restrict__ gamma,
    const float* __restrict__ beta, float* __restrict__ stats)
{
    const int o = blockIdx.x;
    float s = 0.0f, ss = 0.0f;
    for (int b = 0; b < BATCH; ++b) {
        const float* p = pre + ((size_t)b * COUT + o) * HWSZ;
        for (int i = threadIdx.x; i < HWSZ; i += 256) {
            float v = p[i];
            s += v; ss += v * v;
        }
    }
#pragma unroll
    for (int d = 16; d > 0; d >>= 1) {
        s  += __shfl_down(s,  d, 32);
        ss += __shfl_down(ss, d, 32);
    }
    __shared__ float rs[8], rss[8];
    if ((threadIdx.x & 31) == 0) {
        rs[threadIdx.x >> 5]  = s;
        rss[threadIdx.x >> 5] = ss;
    }
    __syncthreads();
    if (threadIdx.x == 0) {
        float S = 0.0f, SS = 0.0f;
#pragma unroll
        for (int i = 0; i < 8; ++i) { S += rs[i]; SS += rss[i]; }
        const float invN = 1.0f / (float)(BATCH * HWSZ);
        float mean = S * invN;
        float var  = SS * invN - mean * mean;
        float inv  = gamma[o] * rsqrtf(var + 1e-5f);
        stats[o * 2 + 0] = inv;
        stats[o * 2 + 1] = beta[o] - mean * inv;
    }
}

// ---------------------------------------------------------------------------
// K4: apply BN + ReLU
// ---------------------------------------------------------------------------
__global__ __launch_bounds__(256) void k4_bn_apply(
    const float* __restrict__ pre, const float* __restrict__ stats,
    float* __restrict__ out)
{
    int idx = blockIdx.x * 256 + threadIdx.x;
    int o   = (idx >> 12) & (COUT - 1);
    float v = pre[idx] * stats[o * 2 + 0] + stats[o * 2 + 1];
    out[idx] = fmaxf(v, 0.0f);
}

// ---------------------------------------------------------------------------
extern "C" void kernel_launch(void* const* d_in, const int* in_sizes, int n_in,
                              void* d_out, int out_size, void* d_ws, size_t ws_size,
                              hipStream_t stream)
{
    (void)in_sizes; (void)n_in; (void)out_size; (void)ws_size;
    const float* x      = (const float*)d_in[0];
    const float* w_off  = (const float*)d_in[1];
    const float* b_off  = (const float*)d_in[2];
    const float* w_conv = (const float*)d_in[3];
    const float* gamma  = (const float*)d_in[4];
    const float* beta   = (const float*)d_in[5];
    float* out = (float*)d_out;

    const size_t off_elems = (size_t)BATCH * OFFCH * HWSZ;   // 294912
    const size_t pre_elems = (size_t)BATCH * COUT * HWSZ;    // 4194304
    const size_t w_elems   = (size_t)KTAPS * COUT * CIN;     // 589824

    float*  offb  = (float*)d_ws;
    float*  pre   = offb + off_elems;
    float*  stats = pre + pre_elems;
    __bf16* wh    = (__bf16*)(stats + 512);
    __bf16* wl    = wh + w_elems;

    (void)hipFuncSetAttribute((const void*)k2_deform_gemm,
                              hipFuncAttributeMaxDynamicSharedMemorySize,
                              SMEM_BYTES);

    k0_wsplit<<<dim3((unsigned)(w_elems / 256)), dim3(256), 0, stream>>>(w_conv, wh, wl);
    k1_offset_conv<<<dim3(BATCH * HH), dim3(64), 0, stream>>>(x, w_off, b_off, offb);
    k2_deform_gemm<<<dim3(COUT / 128, BATCH * HH), dim3(256), SMEM_BYTES, stream>>>(
        x, wh, wl, offb, pre);
    k3_bn_stats<<<dim3(COUT), dim3(256), 0, stream>>>(pre, gamma, beta, stats);
    k4_bn_apply<<<dim3((unsigned)(pre_elems / 256)), dim3(256), 0, stream>>>(pre, stats, out);
}
